// CnfGraphModel_59150289601136
// MI455X (gfx1250) — matrix operand order repaired and checked by reading, not compiled, not run on hardware
//
#include <hip/hip_runtime.h>
#include <hip/hip_bf16.h>
#include <cstdint>
#include <cstddef>

#define NV_ 20000
#define NR_ 40000
#define NI_ 40000
#define N_  100000
#define E_  800000
#define HD  128

typedef __attribute__((ext_vector_type(16))) _Float16 v16h;
typedef __attribute__((ext_vector_type(8)))  float    v8f;
typedef __attribute__((ext_vector_type(4)))  _Float16 h4;
typedef __attribute__((ext_vector_type(4)))  float    f4;

// ---------------------------------------------------------------------------
// TDM (Tensor Data Mover) support: this toolchain exposes the 6-arg builtin
// (uint32x4 g0, int32x8 g1, int32x4 g2, int32x4 g3, int32x8 extra, i32 cpol).
// ---------------------------------------------------------------------------
#ifdef __has_builtin
#if __has_builtin(__builtin_amdgcn_tensor_load_to_lds) && \
    __has_builtin(__builtin_amdgcn_s_wait_tensorcnt)
#define HAVE_TDM 1
#endif
#endif

#ifdef HAVE_TDM
typedef __attribute__((ext_vector_type(4))) unsigned int u32x4;
typedef __attribute__((ext_vector_type(8))) int          i32x8;
typedef __attribute__((ext_vector_type(4))) int          i32x4;

// DMA 32 KB (4096 x 8B) from global into LDS via one TDM descriptor.
// D# group0: count=1, lds_addr, global_addr[56:0], type=2 ("image").
// D# group1: data_size=3 (8B), tensor_dim0=tile_dim0=4096, tensor_dim1=1,
//            dim0_stride=4096, no pad / no multicast / no barrier.
__device__ __forceinline__ void tdm_load_b_to_lds(const _Float16* gsrc, unsigned ldsoff) {
    unsigned long long ga = (unsigned long long)(size_t)gsrc;
    const unsigned n8 = 4096u; // 32KB in 8-byte units
    u32x4 g0 = (u32x4){1u,                                     // count=1
                       ldsoff,                                 // lds_addr
                       (unsigned)ga,                           // global_addr[31:0]
                       ((unsigned)(ga >> 32) & 0x01FFFFFFu) | 0x80000000u}; // [56:32] | type=2
    i32x8 g1;
    g1[0] = 0x00030000;                          // data_size=3 (8 bytes)
    g1[1] = (int)((n8 & 0xFFFFu) << 16);         // tensor_dim0[15:0]
    g1[2] = (int)(((n8 >> 16) & 0xFFFFu) | (1u << 16)); // tensor_dim0 hi | tensor_dim1=1
    g1[3] = (int)((n8 & 0xFFFFu) << 16);         // tile_dim0 = 4096
    g1[4] = 1;                                   // tile_dim1 = 1
    g1[5] = (int)n8;                             // tensor_dim0_stride = 4096
    g1[6] = 0;
    g1[7] = 0;
    i32x4 z4 = (i32x4){0, 0, 0, 0};
    i32x8 z8;
    z8[0] = 0; z8[1] = 0; z8[2] = 0; z8[3] = 0;
    z8[4] = 0; z8[5] = 0; z8[6] = 0; z8[7] = 0;
    __builtin_amdgcn_tensor_load_to_lds(g0, g1, z4, z4, z8, 0);
}
#endif

// ---------------------------------------------------------------------------
// float atomic max via integer tricks (safe, init must be -inf)
// ---------------------------------------------------------------------------
__device__ __forceinline__ void atomicMaxF(float* addr, float val) {
    if (val >= 0.0f) atomicMax((int*)addr, __float_as_int(val));
    else             atomicMin((unsigned int*)addr, __float_as_uint(val));
}

__device__ __forceinline__ float waveRed32(float v) {
#pragma unroll
    for (int o = 16; o > 0; o >>= 1) v += __shfl_xor(v, o, 32);
    return v;
}

// ---------------------------------------------------------------------------
// Pack a 128x128 fp32 weight (K rows, N cols) into WMMA B-fragment order fp16.
// Layout per (ntile, ktile): 32 lanes x 16 halves; lane L: n = nt*16+(L&15),
// hi = L>>4; element e: p=e>>1, hh=e&1, k = kt*32 + hi*16 + 2p + hh.
// ---------------------------------------------------------------------------
__global__ void pack_b_kernel(const float* __restrict__ W, _Float16* __restrict__ out) {
    int idx = blockIdx.x * blockDim.x + threadIdx.x;
    if (idx >= 8 * 4 * 32 * 16) return;
    int e    = idx & 15;
    int lane = (idx >> 4) & 31;
    int kt   = (idx >> 9) & 3;
    int nt   = idx >> 11;
    int p = e >> 1, hh = e & 1, hi = lane >> 4;
    int n = nt * 16 + (lane & 15);
    int k = kt * 32 + hi * 16 + 2 * p + hh;
    out[idx] = (_Float16)W[k * HD + n];
}

// ---------------------------------------------------------------------------
// WMMA GEMM with TDM-staged B in LDS (uniform weight set for all rows).
// One wave per 16-row tile, 8 waves/WG; B staged once per workgroup.
// C(fp16)[nrows,128] = A(fp32)[nrows,128] @ W + bias
// ---------------------------------------------------------------------------
__global__ __launch_bounds__(256)
void gemm_wmma_lds_kernel(const float* __restrict__ A,
                          const _Float16* __restrict__ Bp,
                          const float* __restrict__ bias,
                          _Float16* __restrict__ C, int nrows) {
    __shared__ __align__(32) _Float16 Bsm[HD * HD]; // 32 KB packed B fragments

#ifdef HAVE_TDM
    if ((threadIdx.x >> 5) == 0) {                 // one wave issues the DMA
        tdm_load_b_to_lds(Bp, (unsigned)(size_t)(&Bsm[0]));
        __builtin_amdgcn_s_wait_tensorcnt(0);
    }
#else
    {
        const uint4* gs = (const uint4*)Bp;
        uint4*       ls = (uint4*)Bsm;
        for (int j = threadIdx.x; j < (HD * HD * 2) / 16; j += 256) ls[j] = gs[j];
    }
#endif
    __syncthreads();

    int wave = (blockIdx.x * blockDim.x + threadIdx.x) >> 5;
    int lane = threadIdx.x & 31;
    int row0 = wave * 16;
    if (row0 >= nrows) return;

    int r  = row0 + (lane & 15);
    int hi = lane >> 4;
    const float* Arow = A + (size_t)r * HD;

    v8f acc[8];
#pragma unroll
    for (int nt = 0; nt < 8; ++nt) acc[nt] = (v8f){0.f,0.f,0.f,0.f,0.f,0.f,0.f,0.f};

#pragma unroll
    for (int kt = 0; kt < 4; ++kt) {
        v16h a;
#pragma unroll
        for (int p = 0; p < 8; ++p) {
            int kk = kt * 32 + hi * 16 + 2 * p;
            float2 f = *(const float2*)(Arow + kk);
            a[2 * p]     = (_Float16)f.x;
            a[2 * p + 1] = (_Float16)f.y;
        }
#pragma unroll
        for (int nt = 0; nt < 8; ++nt) {
            v16h b = *(const v16h*)(Bsm + ((((nt * 4 + kt) * 32) + lane) << 4));
            acc[nt] = __builtin_amdgcn_wmma_f32_16x16x32_f16(
                false, a, false, b, (short)0, acc[nt], false, false);
        }
    }

    int n = lane & 15;
#pragma unroll
    for (int nt = 0; nt < 8; ++nt) {
        float bv = bias[nt * 16 + n];
#pragma unroll
        for (int v = 0; v < 8; ++v) {
            int m = hi * 8 + v;
            C[(size_t)(row0 + m) * HD + nt * 16 + n] = (_Float16)(acc[nt][v] + bv);
        }
    }
}

// ---------------------------------------------------------------------------
// WMMA GEMM, direct-global B, weight set selected by node-type region of the
// row tile (regions are 16-aligned; a WG may straddle, so per-wave select).
// Used for the residual projection (fp32 out).
// ---------------------------------------------------------------------------
__global__ __launch_bounds__(256)
void gemm_wmma_res_kernel(const float* __restrict__ A,
                          const _Float16* __restrict__ B0,
                          const _Float16* __restrict__ B1,
                          const _Float16* __restrict__ B2,
                          const float* __restrict__ bias0,
                          const float* __restrict__ bias1,
                          const float* __restrict__ bias2,
                          float* __restrict__ C, int nrows) {
    int wave = (blockIdx.x * blockDim.x + threadIdx.x) >> 5;
    int lane = threadIdx.x & 31;
    int row0 = wave * 16;
    if (row0 >= nrows) return;

    const _Float16* Bp = B0;
    const float*    bb = bias0;
    if (row0 >= NV_ + NR_)      { Bp = B2; bb = bias2; }
    else if (row0 >= NV_)       { Bp = B1; bb = bias1; }

    int r  = row0 + (lane & 15);
    int hi = lane >> 4;
    const float* Arow = A + (size_t)r * HD;

    v8f acc[8];
#pragma unroll
    for (int nt = 0; nt < 8; ++nt) acc[nt] = (v8f){0.f,0.f,0.f,0.f,0.f,0.f,0.f,0.f};

#pragma unroll
    for (int kt = 0; kt < 4; ++kt) {
        v16h a;
#pragma unroll
        for (int p = 0; p < 8; ++p) {
            int kk = kt * 32 + hi * 16 + 2 * p;
            float2 f = *(const float2*)(Arow + kk);
            a[2 * p]     = (_Float16)f.x;
            a[2 * p + 1] = (_Float16)f.y;
        }
#pragma unroll
        for (int nt = 0; nt < 8; ++nt) {
            v16h b = *(const v16h*)(Bp + ((((nt * 4 + kt) * 32) + lane) << 4));
            acc[nt] = __builtin_amdgcn_wmma_f32_16x16x32_f16(
                false, a, false, b, (short)0, acc[nt], false, false);
        }
    }

    int n = lane & 15;
#pragma unroll
    for (int nt = 0; nt < 8; ++nt) {
        float bv = bb[nt * 16 + n];
#pragma unroll
        for (int v = 0; v < 8; ++v) {
            int m = hi * 8 + v;
            C[(size_t)(row0 + m) * HD + nt * 16 + n] = acc[nt][v] + bv;
        }
    }
}

// ---------------------------------------------------------------------------
// Encoder: h[i,c] = silu( concat(global, x_feat) @ (Wv|Wc) + (bv|bc) )
// ---------------------------------------------------------------------------
__global__ void encoder_kernel(const float* __restrict__ x,
                               const float* __restrict__ gdata,
                               const float* __restrict__ Wv, const float* __restrict__ bv,
                               const float* __restrict__ Wc, const float* __restrict__ bc,
                               float* __restrict__ h) {
    int idx = blockIdx.x * blockDim.x + threadIdx.x;
    if (idx >= N_ * HD) return;
    int i = idx >> 7, c = idx & 127;
    const float* W = (i < NV_) ? Wv : Wc;
    const float* B = (i < NV_) ? bv : bc;
    float acc = B[c];
#pragma unroll
    for (int j = 0; j < 16; ++j) acc += gdata[j] * W[j * HD + c];
    const float* xr = x + (size_t)i * 8;
#pragma unroll
    for (int j = 0; j < 8; ++j) acc += xr[j] * W[(16 + j) * HD + c];
    h[idx] = acc / (1.0f + __expf(-acc)); // silu
}

// ---------------------------------------------------------------------------
struct EdgeP {
    const _Float16* XL[3];
    const _Float16* XR[3];
    const float*    We[3];
    const float*    att[3];
};

__global__ void init_nodes_kernel(float* __restrict__ agg,
                                  float* __restrict__ smax,
                                  float* __restrict__ den) {
    int idx = blockIdx.x * blockDim.x + threadIdx.x;
    if (idx >= N_ * HD) return;
    agg[idx] = 0.0f;
    if (idx < N_) { smax[idx] = __int_as_float(0xFF800000); den[idx] = 0.0f; }
}

// pass 1: per-edge attention score + segment max (wave per edge, 4 ch/lane)
__global__ void edge_score_kernel(const int* __restrict__ src,
                                  const int* __restrict__ dst,
                                  const int* __restrict__ ntype,
                                  const float* __restrict__ eattr,
                                  EdgeP P,
                                  float* __restrict__ score,
                                  float* __restrict__ smax) {
    int w = (blockIdx.x * blockDim.x + threadIdx.x) >> 5;
    if (w >= E_) return;
    int lane = threadIdx.x & 31;
    int s = src[w], d = dst[w];
    int t = ntype[d];
    const _Float16* xl = P.XL[t] + (size_t)s * HD;
    const _Float16* xr = P.XR[t] + (size_t)d * HD;
    float ea = eattr[w];
    int c0 = lane * 4;
    h4 av = *(const h4*)(xl + c0);
    h4 bv = *(const h4*)(xr + c0);
    f4 we = *(const f4*)(P.We[t] + c0);
    f4 at = *(const f4*)(P.att[t] + c0);
    float partial = 0.0f;
#pragma unroll
    for (int j = 0; j < 4; ++j) {
        float m  = (float)av[j] + (float)bv[j] + ea * we[j];
        float lr = (m > 0.0f) ? m : 0.2f * m;
        partial += lr * at[j];
    }
    partial = waveRed32(partial);
    if (lane == 0) {
        score[w] = partial;
        atomicMaxF(smax + d, partial);
    }
}

// pass 2: w = exp(score - smax[dst]); den[dst] += w
__global__ void edge_norm_kernel(const int* __restrict__ dst,
                                 const float* __restrict__ score,
                                 const float* __restrict__ smax,
                                 float* __restrict__ wbuf,
                                 float* __restrict__ den) {
    int e = blockIdx.x * blockDim.x + threadIdx.x;
    if (e >= E_) return;
    int d = dst[e];
    float w = __expf(score[e] - smax[d]);
    wbuf[e] = w;
    atomicAdd(den + d, w);
}

// pass 3: agg[dst] += alpha * xl[src]
__global__ void edge_agg_kernel(const int* __restrict__ src,
                                const int* __restrict__ dst,
                                const int* __restrict__ ntype,
                                EdgeP P,
                                const float* __restrict__ wbuf,
                                const float* __restrict__ den,
                                float* __restrict__ agg) {
    int w = (blockIdx.x * blockDim.x + threadIdx.x) >> 5;
    if (w >= E_) return;
    int lane = threadIdx.x & 31;
    int s = src[w], d = dst[w];
    int t = ntype[d];
    float alpha = wbuf[w] / (den[d] + 1e-16f);
    const _Float16* xl = P.XL[t] + (size_t)s * HD;
    int c0 = lane * 4;
    h4 av = *(const h4*)(xl + c0);
    float* ap = agg + (size_t)d * HD + c0;
#pragma unroll
    for (int j = 0; j < 4; ++j) atomicAdd(ap + j, alpha * (float)av[j]);
}

__global__ void finalize_kernel(const float* __restrict__ agg,
                                const float* __restrict__ xres,
                                float* __restrict__ h) {
    int idx = blockIdx.x * blockDim.x + threadIdx.x;
    if (idx >= N_ * HD) return;
    h[idx] = fmaxf(agg[idx] + xres[idx], 0.0f);
}

// ---------------------------------------------------------------------------
// Heads
// ---------------------------------------------------------------------------
__global__ void init_scal_kernel(float* scal) {
    if (threadIdx.x < 8) scal[threadIdx.x] = 0.0f;
}

__global__ void logits_kernel(const float* __restrict__ h,
                              const float* __restrict__ Wout,
                              const float* __restrict__ bout,
                              float* __restrict__ logit,
                              float* __restrict__ scal) {
    int w = (blockIdx.x * blockDim.x + threadIdx.x) >> 5;
    if (w >= NR_) return;
    int lane = threadIdx.x & 31;
    const float* hr = h + (size_t)(NV_ + w) * HD;
    int c0 = lane * 4;
    f4 hv = *(const f4*)(hr + c0);
    f4 wv = *(const f4*)(Wout + c0);
    float partial = hv[0]*wv[0] + hv[1]*wv[1] + hv[2]*wv[2] + hv[3]*wv[3];
    partial = waveRed32(partial);
    if (lane == 0) {
        float l = partial + bout[0];
        logit[w] = l;
        atomicAdd(scal + 0, l);
        atomicAdd(scal + 1, l * l);
    }
}

__global__ void normalize_kernel(const float* __restrict__ logit,
                                 const float* __restrict__ scal,
                                 float* __restrict__ logp) {
    int i = blockIdx.x * blockDim.x + threadIdx.x;
    if (i >= NR_) return;
    float sum = scal[0], sumsq = scal[1];
    float mean = sum / (float)NR_;
    float var  = (sumsq - sum * mean) / (float)(NR_ - 1);
    float sd   = sqrtf(var);
    float z = (logit[i] - mean) / sd;
    float p = 1.0f / (1.0f + __expf(-z));
    p = fminf(fmaxf(p, 1e-20f), 1.0f - 1e-7f);
    logp[i] = __logf(p);
}

// ExpectedValueNormalization: 10 Newton steps on gamma, single workgroup
__global__ void evn_kernel(const float* __restrict__ logp,
                           float* __restrict__ out, float ex) {
    __shared__ float sf[512], sfp[512], sg;
    int tid = threadIdx.x;
    float gamma = 1.0f;
    for (int it = 0; it < 10; ++it) {
        float f = 0.0f, fp = 0.0f;
        for (int i = tid; i < NR_; i += 512) {
            float lp = logp[i];
            float pg = __expf(gamma * lp);
            f  += pg;
            fp += pg * lp;
        }
        sf[tid] = f; sfp[tid] = fp;
        __syncthreads();
        for (int s = 256; s > 0; s >>= 1) {
            if (tid < s) { sf[tid] += sf[tid + s]; sfp[tid] += sfp[tid + s]; }
            __syncthreads();
        }
        if (tid == 0) {
            float g2 = gamma - (sf[0] - ex) / sfp[0];
            sg = fmaxf(g2, gamma * 0.1f);
        }
        __syncthreads();
        gamma = sg;
        __syncthreads();
    }
    for (int i = tid; i < NR_; i += 512) out[i] = __expf(gamma * logp[i]);
}

__global__ void value_kernel(const float* __restrict__ h,
                             const float* __restrict__ Wvv, const float* __restrict__ bvv,
                             const float* __restrict__ Wvc, const float* __restrict__ bvc,
                             float* __restrict__ scal) {
    int w = (blockIdx.x * blockDim.x + threadIdx.x) >> 5;
    if (w >= N_) return;
    int lane = threadIdx.x & 31;
    const float* Wp = (w < NV_) ? Wvv : Wvc;
    const float* bp = (w < NV_) ? bvv : bvc;
    int c0 = lane * 4;
    f4 hv = *(const f4*)(h + (size_t)w * HD + c0);
    f4 wv = *(const f4*)(Wp + c0);
    float partial = hv[0]*wv[0] + hv[1]*wv[1] + hv[2]*wv[2] + hv[3]*wv[3];
    partial = waveRed32(partial);
    if (lane == 0) atomicAdd(scal + 2, partial + bp[0]);
}

__global__ void final_value_kernel(const float* __restrict__ scal, float* __restrict__ out) {
    out[NR_] = scal[2] / (float)N_;
}

// ---------------------------------------------------------------------------
// Host launch
// ---------------------------------------------------------------------------
extern "C" void kernel_launch(void* const* d_in, const int* in_sizes, int n_in,
                              void* d_out, int out_size, void* d_ws, size_t ws_size,
                              hipStream_t stream) {
    auto F = [&](int i) { return (const float*)d_in[i]; };
    auto I = [&](int i) { return (const int*)d_in[i]; };

    const float *x = nullptr, *gdata = nullptr, *eattr = nullptr;
    const int *eidx = nullptr, *ntype = nullptr;
    const float *Wv=0,*bv=0,*Wc=0,*bc=0,*Wout=0,*bout=0,*Wvv=0,*bvv=0,*Wvc=0,*bvc=0;
    struct G { const float *Wl,*bl,*Wr,*br,*We,*att,*Wres,*bias; } g[2][3];

    bool sortedTop = (n_in >= 2 && in_sizes[1] == 2 * E_);
    auto fillSortedParams = [&](int base) {
        Wc = F(base + 0); Wout = F(base + 1); Wv = F(base + 2);
        Wvc = F(base + 3); Wvv = F(base + 4); bc = F(base + 5);
        int k = base + 6; // 'blocks' (sorted after 'bc', before 'bout')
        for (int b = 0; b < 2; ++b)
            for (int ts = 0; ts < 3; ++ts) { // sorted keys: irr, red, vars
                int t = 2 - ts;
                g[b][t].We = F(k++); g[b][t].Wl = F(k++); g[b][t].Wr = F(k++);
                g[b][t].Wres = F(k++); g[b][t].att = F(k++); g[b][t].bias = F(k++);
                g[b][t].bl = F(k++); g[b][t].br = F(k++);
            }
        bout = F(k++); bv = F(k++); bvc = F(k++); bvv = F(k++);
    };

    if (!sortedTop) {
        // top-level insertion order: x, global, edge_index, edge_attr, node_type, params..., ex
        x = F(0); gdata = F(1); eidx = I(2); eattr = F(3); ntype = I(4);
        bool insertParams = (n_in > 10 && in_sizes[10] == 1); // bout(1) vs bc(128)
        if (insertParams) {
            Wv=F(5); bv=F(6); Wc=F(7); bc=F(8); Wout=F(9); bout=F(10);
            Wvv=F(11); bvv=F(12); Wvc=F(13); bvc=F(14);
            int k = 15;
            for (int b = 0; b < 2; ++b)
                for (int t = 0; t < 3; ++t) { // vars, red, irr
                    g[b][t].Wl = F(k++); g[b][t].bl = F(k++);
                    g[b][t].Wr = F(k++); g[b][t].br = F(k++);
                    g[b][t].We = F(k++); g[b][t].att = F(k++);
                    g[b][t].Wres = F(k++); g[b][t].bias = F(k++);
                }
        } else {
            fillSortedParams(5);
        }
    } else {
        // fully key-sorted: edge_attr, edge_index, ex, global, node_type, params..., x
        eattr = F(0); eidx = I(1); gdata = F(3); ntype = I(4);
        fillSortedParams(5);
        x = F(63);
    }
    const int* src = eidx;
    const int* dst = eidx + E_;

    // ---- workspace carve ----
    char* wsb = (char*)d_ws;
    size_t off = 0;
    auto take = [&](size_t bytes) -> void* {
        void* p = wsb + off;
        off = (off + bytes + 511) & ~(size_t)511;
        return p;
    };
    float*     h    = (float*)take((size_t)N_ * HD * 4);
    _Float16*  XL[3], *XR[3];
    for (int t = 0; t < 3; ++t) XL[t] = (_Float16*)take((size_t)N_ * HD * 2);
    for (int t = 0; t < 3; ++t) XR[t] = (_Float16*)take((size_t)N_ * HD * 2);
    float*     XRES = (float*)take((size_t)N_ * HD * 4);
    float*     AGG  = (float*)take((size_t)N_ * HD * 4);
    float*     smax = (float*)take((size_t)N_ * 4);
    float*     den  = (float*)take((size_t)N_ * 4);
    float*     scoreB = (float*)take((size_t)E_ * 4);
    float*     wbuf   = (float*)take((size_t)E_ * 4);
    _Float16*  packW  = (_Float16*)take((size_t)18 * 16384 * 2);
    float*     logit  = (float*)take((size_t)NR_ * 4);
    float*     logp   = (float*)take((size_t)NR_ * 4);
    float*     scal   = (float*)take(64);
    (void)ws_size; (void)out_size;

    auto pW = [&](int b, int t, int which) {
        return packW + ((size_t)((b * 3 + t) * 3 + which)) * 16384;
    };

    // ---- pack weights to WMMA B fragments (fp16) ----
    for (int b = 0; b < 2; ++b)
        for (int t = 0; t < 3; ++t) {
            pack_b_kernel<<<64, 256, 0, stream>>>(g[b][t].Wl,   pW(b, t, 0));
            pack_b_kernel<<<64, 256, 0, stream>>>(g[b][t].Wr,   pW(b, t, 1));
            pack_b_kernel<<<64, 256, 0, stream>>>(g[b][t].Wres, pW(b, t, 2));
        }

    const int gridNHD = (N_ * HD + 255) / 256;
    const int gemmGrid = ((N_ / 16) + 7) / 8; // 8 waves (16-row tiles) per WG

    // ---- encoder ----
    encoder_kernel<<<gridNHD, 256, 0, stream>>>(x, gdata, Wv, bv, Wc, bc, h);

    // ---- GAT blocks ----
    for (int b = 0; b < 2; ++b) {
        for (int t = 0; t < 3; ++t) {
            gemm_wmma_lds_kernel<<<gemmGrid, 256, 0, stream>>>(
                h, pW(b, t, 0), g[b][t].bl, XL[t], N_);
            gemm_wmma_lds_kernel<<<gemmGrid, 256, 0, stream>>>(
                h, pW(b, t, 1), g[b][t].br, XR[t], N_);
        }
        gemm_wmma_res_kernel<<<gemmGrid, 256, 0, stream>>>(
            h, pW(b, 0, 2), pW(b, 1, 2), pW(b, 2, 2),
            g[b][0].bias, g[b][1].bias, g[b][2].bias, XRES, N_);

        init_nodes_kernel<<<gridNHD, 256, 0, stream>>>(AGG, smax, den);

        EdgeP ep;
        for (int t = 0; t < 3; ++t) {
            ep.XL[t] = XL[t]; ep.XR[t] = XR[t];
            ep.We[t] = g[b][t].We; ep.att[t] = g[b][t].att;
        }
        edge_score_kernel<<<E_ * 32 / 256, 256, 0, stream>>>(src, dst, ntype, eattr,
                                                             ep, scoreB, smax);
        edge_norm_kernel<<<(E_ + 255) / 256, 256, 0, stream>>>(dst, scoreB, smax, wbuf, den);
        edge_agg_kernel<<<E_ * 32 / 256, 256, 0, stream>>>(src, dst, ntype, ep, wbuf, den, AGG);
        finalize_kernel<<<gridNHD, 256, 0, stream>>>(AGG, XRES, h);
    }

    // ---- heads ----
    float* out = (float*)d_out;
    init_scal_kernel<<<1, 32, 0, stream>>>(scal);
    logits_kernel<<<NR_ * 32 / 256, 256, 0, stream>>>(h, Wout, bout, logit, scal);
    normalize_kernel<<<(NR_ + 255) / 256, 256, 0, stream>>>(logit, scal, logp);
    evn_kernel<<<1, 512, 0, stream>>>(logp, out, 50.0f);
    value_kernel<<<N_ * 32 / 256, 256, 0, stream>>>(h, Wvv, bvv, Wvc, bvc, scal);
    final_value_kernel<<<1, 1, 0, stream>>>(scal, out);
}
